// Linear_extractor_cluster_28449863369095
// MI455X (gfx1250) — matrix operand order repaired and checked by reading, not compiled, not run on hardware
//
#include <hip/hip_runtime.h>
#include <hip/hip_bf16.h>

typedef __attribute__((ext_vector_type(16))) __bf16 v16bf;
typedef __attribute__((ext_vector_type(8)))  float  v8f;

#define B_  512
#define S_  512
#define C_  64
#define D_  512
#define H_  256
#define E_  4
#define KC  64
#define MT  128   // block M (D-dim) tile: 8 waves x 16 rows, per-expert accum
#define LSTR 72   // LDS row stride in bf16 elems: 144B = 16B-aligned, 36-dword

// round-to-nearest-even fp32 -> bf16 (bit pattern as u16)
__device__ __forceinline__ unsigned short f2bf(float f) {
    unsigned u = __float_as_uint(f);
    u += 0x7fffu + ((u >> 16) & 1u);
    return (unsigned short)(u >> 16);
}
__device__ __forceinline__ unsigned pack2bf(float a, float b) {
    return (unsigned)f2bf(a) | ((unsigned)f2bf(b) << 16);
}

// ---------------- prep: expert_w fp32 -> bf16 (layout preserved) ----------
__global__ __launch_bounds__(256) void wconv_kernel(const float* __restrict__ w,
                                                    unsigned short* __restrict__ wbf) {
    const size_t i = ((size_t)blockIdx.x * 256 + threadIdx.x) * 4;   // elem idx
    const float4 v = *(const float4*)(w + i);
    uint2 p;
    p.x = pack2bf(v.x, v.y);
    p.y = pack2bf(v.z, v.w);
    *(uint2*)((unsigned*)wbf + (i >> 1)) = p;
}

// ---- gating: mean over C (fused), h=relu(m@W1^T), logits, top-2 softmax ----
__global__ __launch_bounds__(256) void gate_kernel(const float* __restrict__ x,
                                                   const float* __restrict__ gw1,
                                                   const float* __restrict__ gw2,
                                                   int* __restrict__ eidx,
                                                   float* __restrict__ gval) {
    __shared__ float sm[S_];
    __shared__ float sh[H_];
    __shared__ float slog[E_];
    const int b = blockIdx.x, tid = threadIdx.x;
    for (int s = tid; s < S_; s += 256) {
        const float4* p = (const float4*)(x + ((size_t)b * S_ + s) * C_);
        float sum = 0.f;
        #pragma unroll
        for (int i = 0; i < C_ / 4; ++i) {
            float4 v = p[i];
            sum += v.x + v.y + v.z + v.w;
        }
        sm[s] = sum * (1.0f / C_);
    }
    __syncthreads();
    {
        const float* wr = gw1 + (size_t)tid * S_;
        float acc = 0.f;
        for (int s = 0; s < S_; ++s) acc = fmaf(sm[s], wr[s], acc);
        sh[tid] = acc > 0.f ? acc : 0.f;
    }
    __syncthreads();
    if (tid < E_) {
        const float* w2 = gw2 + (size_t)tid * H_;
        float l = 0.f;
        for (int hh = 0; hh < H_; ++hh) l = fmaf(sh[hh], w2[hh], l);
        slog[tid] = l;
    }
    __syncthreads();
    if (tid == 0) {
        int i0 = 0; float l0 = slog[0];
        #pragma unroll
        for (int e = 1; e < E_; ++e) if (slog[e] > l0) { l0 = slog[e]; i0 = e; }
        int i1 = -1; float l1 = -3.4e38f;
        #pragma unroll
        for (int e = 0; e < E_; ++e)
            if (e != i0 && slog[e] > l1) { l1 = slog[e]; i1 = e; }
        float e1x = __expf(l1 - l0);
        float inv = 1.0f / (1.0f + e1x);
        eidx[2 * b] = i0;  eidx[2 * b + 1] = i1;
        gval[2 * b] = inv; gval[2 * b + 1] = e1x * inv;
    }
}

// ---------------- balance loss: cv^2(importance) + cv^2(load) -------------
__device__ __forceinline__ float cv2(const float v[E_]) {
    float mean = 0.f;
    for (int e = 0; e < E_; ++e) mean += v[e];
    mean *= (1.0f / E_);
    float var = 0.f;
    for (int e = 0; e < E_; ++e) { float d = v[e] - mean; var += d * d; }
    var *= (1.0f / (E_ - 1));              // ddof=1
    return var / (mean * mean + 1e-10f);
}

__global__ __launch_bounds__(256) void loss_kernel(const int* __restrict__ eidx,
                                                   const float* __restrict__ gval,
                                                   float* __restrict__ out) {
    __shared__ float imp[E_];
    __shared__ int   cnt[E_];
    const int tid = threadIdx.x;
    if (tid < E_) { imp[tid] = 0.f; cnt[tid] = 0; }
    __syncthreads();
    for (int b = tid; b < B_; b += 256) {
        atomicAdd(&imp[eidx[2 * b]],     gval[2 * b]);
        atomicAdd(&imp[eidx[2 * b + 1]], gval[2 * b + 1]);
        atomicAdd(&cnt[eidx[2 * b]],     1);
        atomicAdd(&cnt[eidx[2 * b + 1]], 1);
    }
    __syncthreads();
    if (tid == 0) {
        float a[E_], c[E_];
        for (int e = 0; e < E_; ++e) { a[e] = imp[e]; c[e] = (float)cnt[e]; }
        out[(size_t)B_ * D_ * C_] = cv2(a) + cv2(c);
    }
}

// ------- main: y[b] = g0*(W_e0 @ x[b]) + g1*(W_e1 @ x[b]) via WMMA --------
// Block tile 128(D) x 64(C); double-buffered LDS: weights via async DMA,
// x via convert/transpose; per-expert f32 accumulators, gates in epilogue.
__global__ __launch_bounds__(256) void moe_gemm_kernel(
    const float* __restrict__ x, const unsigned short* __restrict__ wbf,
    const int* __restrict__ eidx, const float* __restrict__ gval,
    float* __restrict__ y) {
    __shared__ unsigned short wT[2][2][MT][LSTR];  // [buf][expert][d][s]
    __shared__ unsigned short xT[2][C_][LSTR];     // [buf][c][s]

    const int b   = blockIdx.y;
    const int dt  = blockIdx.x;              // 128-row D tile (0..3)
    const int tid = threadIdx.x;
    const int lane = tid & 31, wave = tid >> 5;

    const int   e0 = eidx[2 * b],  e1 = eidx[2 * b + 1];
    const float g0 = gval[2 * b],  g1 = gval[2 * b + 1];
    const unsigned short* Wb[2] = {
        wbf + ((size_t)e0 * D_ + dt * MT) * S_,
        wbf + ((size_t)e1 * D_ + dt * MT) * S_ };
    const float* xb = x + (size_t)b * S_ * C_;

    v8f acc[2][4] = {};          // [expert][nt]

    const int ln = lane & 15;    // M (A frag) / N (B frag)
    const int gh = lane >> 4;    // lane-group half

    // async DMA of one weight double-tile into LDS buffer `buf`
    auto issueW = [&](int buf, int s0) {
        #pragma unroll
        for (int it = 0; it < 8; ++it) {
            const int idx = tid + it * 256;      // 0..2047
            const int ex  = idx >> 10;           // expert slot
            const int rem = idx & 1023;
            const int row = rem >> 3;
            const int c16 = rem & 7;             // 16B chunk within 64 elems
            const unsigned short* gp = Wb[ex] + (size_t)row * S_ + s0 + c16 * 8;
            const unsigned ldsOff =
                (unsigned)(uintptr_t)&wT[buf][ex][row][c16 * 8];
            asm volatile("global_load_async_to_lds_b128 %0, %1, off"
                         :: "v"(ldsOff),
                            "v"((unsigned long long)(uintptr_t)gp)
                         : "memory");
        }
    };
    // convert + transpose one x tile into LDS buffer `buf`
    auto stageX = [&](int buf, int s0) {
        #pragma unroll
        for (int it = 0; it < 4; ++it) {
            const int idx  = tid + it * 256;
            const int srow = idx >> 4;
            const int cc   = (idx & 15) << 2;
            const float4 v = *(const float4*)(xb + (size_t)(s0 + srow) * C_ + cc);
            xT[buf][cc + 0][srow] = f2bf(v.x);
            xT[buf][cc + 1][srow] = f2bf(v.y);
            xT[buf][cc + 2][srow] = f2bf(v.z);
            xT[buf][cc + 3][srow] = f2bf(v.w);
        }
    };

    // prologue: fill buffer 0
    issueW(0, 0);
    stageX(0, 0);
    asm volatile("s_wait_asynccnt 0x0" ::: "memory");
    __syncthreads();

    for (int i = 0; i < S_ / KC; ++i) {
        const int buf = i & 1;
        const int s0n = (i + 1) * KC;
        if (s0n < S_) {                      // prefetch next chunk
            issueW(buf ^ 1, s0n);
            stageX(buf ^ 1, s0n);
        }
        // ---- compute chunk i from buffer `buf` ----
        const unsigned* arow0 = (const unsigned*)&wT[buf][0][wave * 16 + ln][0];
        const unsigned* arow1 = (const unsigned*)&wT[buf][1][wave * 16 + ln][0];
        #pragma unroll
        for (int kk = 0; kk < KC; kk += 32) {
            union Frag { v16bf v; uint4 q[2]; };
            // A: 16x32 bf16; half gh holds K = gh*8+[0..7] and 16+gh*8+[0..7]
            Frag A0, A1;
            const int a0 = (kk + gh * 8) >> 1;        // u32 index, 16B aligned
            const int a1 = (kk + 16 + gh * 8) >> 1;
            A0.q[0] = *(const uint4*)(arow0 + a0);
            A0.q[1] = *(const uint4*)(arow0 + a1);
            A1.q[0] = *(const uint4*)(arow1 + a0);
            A1.q[1] = *(const uint4*)(arow1 + a1);
            // B: 32x16 bf16; half gh holds K = gh*8+[0..7] and 16+gh*8+[0..7], column N = ln
            const int bb = (kk + gh * 8) >> 1;
            Frag Bv[4];
            #pragma unroll
            for (int nt = 0; nt < 4; ++nt) {
                const unsigned* brow =
                    (const unsigned*)&xT[buf][nt * 16 + ln][0];
                Bv[nt].q[0] = *(const uint4*)(brow + bb);
                Bv[nt].q[1] = *(const uint4*)(brow + bb + 8);
            }
            #pragma unroll
            for (int nt = 0; nt < 4; ++nt) {
                acc[0][nt] = __builtin_amdgcn_wmma_f32_16x16x32_bf16(
                    false, A0.v, false, Bv[nt].v, (short)0, acc[0][nt], false, false);
                acc[1][nt] = __builtin_amdgcn_wmma_f32_16x16x32_bf16(
                    false, A1.v, false, Bv[nt].v, (short)0, acc[1][nt], false, false);
            }
        }
        asm volatile("s_wait_asynccnt 0x0" ::: "memory");
        __syncthreads();
    }

    // epilogue: y = g0*P_e0 + g1*P_e1 ; D frag: VGPR r -> M = gh*8+r, N = ln
    #pragma unroll
    for (int nt = 0; nt < 4; ++nt) {
        #pragma unroll
        for (int r = 0; r < 8; ++r) {
            const int dg = dt * MT + wave * 16 + gh * 8 + r;
            const int cg = nt * 16 + ln;
            y[((size_t)b * D_ + dg) * C_ + cg] =
                fmaf(g0, acc[0][nt][r], g1 * acc[1][nt][r]);
        }
    }
}

extern "C" void kernel_launch(void* const* d_in, const int* in_sizes, int n_in,
                              void* d_out, int out_size, void* d_ws, size_t ws_size,
                              hipStream_t stream) {
    const float* x        = (const float*)d_in[0];
    const float* gate_w1  = (const float*)d_in[1];
    const float* gate_w2  = (const float*)d_in[2];
    const float* expert_w = (const float*)d_in[3];
    // d_in[4] is k (== 2), baked into the top-2 gate kernel.
    float* out = (float*)d_out;

    // workspace layout
    int*            eidx = (int*)d_ws;                        // B*2 ints
    float*          gval = (float*)(eidx + (size_t)B_ * 2);   // B*2 floats
    unsigned short* wbf  = (unsigned short*)(gval + (size_t)B_ * 2); // E*D*S bf16

    wconv_kernel<<<(E_ * D_ * S_) / (256 * 4), 256, 0, stream>>>(expert_w, wbf);
    gate_kernel<<<B_, 256, 0, stream>>>(x, gate_w1, gate_w2, eidx, gval);
    loss_kernel<<<1, 256, 0, stream>>>(eidx, gval, out);
    moe_gemm_kernel<<<dim3(D_ / MT, B_), 256, 0, stream>>>(x, wbf, eidx, gval, out);
}
